// Head_4028679324187
// MI455X (gfx1250) — matrix operand order repaired and checked by reading, not compiled, not run on hardware
//
#include <hip/hip_runtime.h>

#define BB 16
#define TT 2048
#define CC 512
#define HH 64

typedef _Float16 v16h __attribute__((ext_vector_type(16)));
typedef _Float16 v8h  __attribute__((ext_vector_type(8)));
typedef float    v8f  __attribute__((ext_vector_type(8)));

union AFrag { v16h v; v8h h[2]; };

static __device__ __forceinline__ v8f wmma_f16(v16h a, v16h b, v8f c) {
  // D = A(16x32 f16) x B(32x16 f16) + C(16x16 f32)
  return __builtin_amdgcn_wmma_f32_16x16x32_f16(false, a, false, b, (short)0, c,
                                                false, false);
}

// ---------------------------------------------------------------------------
// Prep: Er -> f16 row-major; Wq/Wk/Wv -> f16 pre-swizzled into B-fragment order
// so the projection kernel loads each B fragment as one contiguous 32B/lane.
// Fragment element e = ((kk*4+hh)*32 + lane)*16 + idx maps to
//   K = kk*32 + (lane>=16 ? 16 : 0) + idx,  N(h) = hh*16 + (lane&15)
// in the contiguous k order the A fragments also use (lanes 0-15: K 0..15, lanes 16-31: K 16..31).
// ---------------------------------------------------------------------------
__global__ void prep_kernel(const float* __restrict__ Wq, const float* __restrict__ Wk,
                            const float* __restrict__ Wv, const float* __restrict__ Er,
                            _Float16* __restrict__ Wswz, _Float16* __restrict__ ErH) {
  const int i = blockIdx.x * blockDim.x + threadIdx.x;
  const int WN = CC * HH;                      // 32768
  if (i < 3 * WN) {
    const int m = i / WN, e = i % WN;
    const float* W = (m == 0) ? Wq : ((m == 1) ? Wk : Wv);
    const int idx  = e & 15;
    const int lane = (e >> 4) & 31;
    const int hh   = (e >> 9) & 3;
    const int kk   = e >> 11;
    const int k = kk * 32 + ((lane >> 4) << 4) + idx;
    const int h = hh * 16 + (lane & 15);
    Wswz[i] = (_Float16)W[k * HH + h];
  } else if (i < 3 * WN + TT * HH) {
    const int e = i - 3 * WN;
    ErH[e] = (_Float16)Er[e];
  }
}

// ---------------------------------------------------------------------------
// Projection: q = x@Wq, k = x@Wk, v = x@Wv (f16 out, fp32 accum via WMMA).
// 4 waves/block, each wave: 16 rows x 64 cols x 3 matrices, K-loop of 16.
// qh, kh stored [B*T, 64] row-major; v stored transposed vT [B*64, T].
// ---------------------------------------------------------------------------
__global__ __launch_bounds__(128)
void proj_kernel(const float* __restrict__ x, const _Float16* __restrict__ Wswz,
                 _Float16* __restrict__ qh, _Float16* __restrict__ kh,
                 _Float16* __restrict__ vT) {
  const int lane = threadIdx.x & 31;
  const int w    = threadIdx.x >> 5;
  const int g    = lane >> 4;      // half-wave group
  const int n    = lane & 15;
  const int rg0  = blockIdx.x * 64 + w * 16;   // global row base (b*T + t)
  const int arow = rg0 + n;                    // A-fragment row for this lane

  v8f acc[12];
#pragma unroll
  for (int i = 0; i < 12; ++i) acc[i] = (v8f){};

  for (int ks = 0; ks < CC / 32; ++ks) {
    const int k0 = ks * 32;
    // A fragment: 16-bit A 16x32. halves[0..15] = K g*16..g*16+15
    // (contiguous, matching the B fragment order).
    const v8f lo = *(const v8f*)(x + (size_t)arow * CC + k0 + g * 16);
    const v8f hi = *(const v8f*)(x + (size_t)arow * CC + k0 + g * 16 + 8);
    AFrag a;
#pragma unroll
    for (int i = 0; i < 8; ++i) { a.v[i] = (_Float16)lo[i]; a.v[8 + i] = (_Float16)hi[i]; }
#pragma unroll
    for (int m = 0; m < 3; ++m)
#pragma unroll
      for (int hh = 0; hh < 4; ++hh) {
        const v16h bfrag = *(const v16h*)(Wswz + m * CC * HH +
                                          (((ks * 4 + hh) * 32 + lane) << 4));
        acc[m * 4 + hh] = wmma_f16(a.v, bfrag, acc[m * 4 + hh]);
      }
  }

  const int b  = rg0 >> 11;      // / T
  const int tl = rg0 & (TT - 1); // t within sequence
#pragma unroll
  for (int hh = 0; hh < 4; ++hh) {
    const int hc = hh * 16 + n;
#pragma unroll
    for (int v = 0; v < 8; ++v) {
      const int m = v + 8 * g;   // C-layout: M = vgpr + 8*(lane/16)
      qh[(size_t)(rg0 + m) * HH + hc] = (_Float16)acc[hh][v];
      kh[(size_t)(rg0 + m) * HH + hc] = (_Float16)acc[4 + hh][v];
    }
    v8h vv;
#pragma unroll
    for (int v = 0; v < 8; ++v) vv[v] = (_Float16)acc[8 + hh][v];
    *(v8h*)(vT + (size_t)(b * HH + hc) * TT + tl + 8 * g) = vv;
  }
}

// ---------------------------------------------------------------------------
// Flash attention with skewed relative bias. 4 waves/block, 16 q-rows/wave,
// 32 keys per iteration. Causal mask + online softmax (fp32 stats).
// skew(rel)[t,s]: diag -> q[t].Er[T-1]; t==s+1 -> 0; t>s+1 -> rel[s+1, t-s-2].
// ---------------------------------------------------------------------------
__global__ __launch_bounds__(128)
void attn_kernel(const _Float16* __restrict__ qh, const _Float16* __restrict__ kh,
                 const _Float16* __restrict__ vT, const _Float16* __restrict__ ErH,
                 float* __restrict__ out) {
  __shared__ float    relbuf[4][16][32];
  __shared__ _Float16 pbuf[4][16][32];
  __shared__ float    diagbuf[4][16];

  const int lane = threadIdx.x & 31;
  const int w    = threadIdx.x >> 5;
  const int g    = lane >> 4;
  const int n    = lane & 15;
  const int b    = blockIdx.x >> 5;            // 32 q-blocks per batch
  const int t0   = (blockIdx.x & 31) * 64 + w * 16;
  const size_t qbase = (size_t)b * TT * HH;

  // Per-row diagonal bias: q[t] . Er[T-1]
  if (lane < 16) {
    float d = 0.f;
    const _Float16* qr = qh + qbase + (size_t)(t0 + lane) * HH;
    const _Float16* er = ErH + (size_t)(TT - 1) * HH;
    for (int h = 0; h < HH; ++h) d += (float)qr[h] * (float)er[h];
    diagbuf[w][lane] = d;
  }
  asm volatile("s_wait_dscnt 0" ::: "memory");
  float dv[8];
#pragma unroll
  for (int v = 0; v < 8; ++v) dv[v] = diagbuf[w][v + 8 * g];

  // Q A-fragments (K halves 0..31 and 32..63), rows t0..t0+15
  AFrag qa0, qa1;
  {
    const _Float16* qr = qh + qbase + (size_t)(t0 + n) * HH;
    qa0.h[0] = *(const v8h*)(qr + g * 16);
    qa0.h[1] = *(const v8h*)(qr + g * 16 + 8);
    qa1.h[0] = *(const v8h*)(qr + 32 + g * 16);
    qa1.h[1] = *(const v8h*)(qr + 32 + g * 16 + 8);
  }

  v8f O[4];
#pragma unroll
  for (int i = 0; i < 4; ++i) O[i] = (v8f){};
  float rowmax[8], rowsum[8];
#pragma unroll
  for (int v = 0; v < 8; ++v) { rowmax[v] = -1e30f; rowsum[v] = 0.f; }

  const float scale = 0.125f;   // 1/sqrt(64)

  for (int j0 = 0; j0 < t0 + 16; j0 += 32) {
    {
      const int pj = (j0 + 32 < TT) ? (j0 + 32) : (TT - 1);
      __builtin_prefetch(kh + qbase + (size_t)pj * HH, 0, 1);     // next K block
      __builtin_prefetch(vT + (size_t)b * HH * TT + pj, 0, 1);    // next V block
    }
    v8f S[2];
#pragma unroll
    for (int st = 0; st < 2; ++st) {
      const int s0 = j0 + st * 16;
      // ---- S = Q @ K^T (B operand: lane=col s, 16 contiguous halves of h)
      int sc = s0 + n; if (sc > TT - 1) sc = TT - 1;
      const _Float16* kr = kh + qbase + (size_t)sc * HH;
      const v16h kf0 = *(const v16h*)(kr + g * 16);
      const v16h kf1 = *(const v16h*)(kr + 32 + g * 16);
      v8f s = wmma_f16(qa0.v, kf0, (v8f){});
      s = wmma_f16(qa1.v, kf1, s);

      // ---- REL tile: rows i = s0+1.., cols r = t0-s0-17 .. +31
      const int i0 = s0 + 1;
      const int r0 = t0 - s0 - 17;
      int ri = i0 + n; if (ri > TT - 1) ri = TT - 1;
      AFrag qs0, qs1;
      const _Float16* qsr = qh + qbase + (size_t)ri * HH;
      qs0.h[0] = *(const v8h*)(qsr + g * 16);
      qs0.h[1] = *(const v8h*)(qsr + g * 16 + 8);
      qs1.h[0] = *(const v8h*)(qsr + 32 + g * 16);
      qs1.h[1] = *(const v8h*)(qsr + 32 + g * 16 + 8);
#pragma unroll
      for (int rc = 0; rc < 2; ++rc) {
        const int rr = r0 + rc * 16 + n;
        v16h e0 = {}, e1 = {};
        if (rr >= 0) {  // rr < T guaranteed
          const _Float16* er = ErH + (size_t)rr * HH;
          e0 = *(const v16h*)(er + g * 16);
          e1 = *(const v16h*)(er + 32 + g * 16);
        }
        v8f R = wmma_f16(qs0.v, e0, (v8f){});
        R = wmma_f16(qs1.v, e1, R);
#pragma unroll
        for (int v = 0; v < 8; ++v) relbuf[w][v + 8 * g][rc * 16 + n] = R[v];
      }
      asm volatile("s_wait_dscnt 0" ::: "memory");

      // ---- combine: scale, skewed rel bias, causal mask
#pragma unroll
      for (int v = 0; v < 8; ++v) {
        const int m = v + 8 * g;
        const int t = t0 + m;
        const int si = s0 + n;
        float val = s[v] * scale;
        if (t == si)          val += dv[v];
        else if (t > si + 1)  val += relbuf[w][n][m - n + 15]; // rel[s+1, t-s-2]
        if (t < si)           val = -1e30f;
        s[v] = val;
      }
      S[st] = s;
    }

    // ---- online softmax (rows live in one 16-lane group, per ISA C-layout)
    float al[8];
#pragma unroll
    for (int v = 0; v < 8; ++v) {
      float c = fmaxf(S[0][v], S[1][v]);
      c = fmaxf(c, __shfl_xor(c, 1));
      c = fmaxf(c, __shfl_xor(c, 2));
      c = fmaxf(c, __shfl_xor(c, 4));
      c = fmaxf(c, __shfl_xor(c, 8));
      const float mnew = fmaxf(rowmax[v], c);
      al[v] = __expf(rowmax[v] - mnew);
      rowmax[v] = mnew;
      const float p0 = __expf(S[0][v] - mnew);
      const float p1 = __expf(S[1][v] - mnew);
      S[0][v] = p0; S[1][v] = p1;
      float rs = p0 + p1;
      rs += __shfl_xor(rs, 1);
      rs += __shfl_xor(rs, 2);
      rs += __shfl_xor(rs, 4);
      rs += __shfl_xor(rs, 8);
      rowsum[v] = rowsum[v] * al[v] + rs;
    }
#pragma unroll
    for (int hh = 0; hh < 4; ++hh)
#pragma unroll
      for (int v = 0; v < 8; ++v) O[hh][v] *= al[v];

    // ---- P -> LDS (f16), re-read as A-fragment for P @ V
#pragma unroll
    for (int v = 0; v < 8; ++v) {
      pbuf[w][v + 8 * g][n]      = (_Float16)S[0][v];
      pbuf[w][v + 8 * g][16 + n] = (_Float16)S[1][v];
    }
    asm volatile("s_wait_dscnt 0" ::: "memory");
    AFrag pa;
    pa.h[0] = *(const v8h*)&pbuf[w][n][g * 16];
    pa.h[1] = *(const v8h*)&pbuf[w][n][g * 16 + 8];

    int js = j0 + g * 16; if (js > TT - 16) js = TT - 16;  // masked cols: P==0
#pragma unroll
    for (int hh = 0; hh < 4; ++hh) {
      const v16h vf = *(const v16h*)(vT + ((size_t)b * HH + hh * 16 + n) * TT + js);
      O[hh] = wmma_f16(pa.v, vf, O[hh]);
    }
  }

  // ---- epilogue: out = O / rowsum
#pragma unroll
  for (int hh = 0; hh < 4; ++hh)
#pragma unroll
    for (int v = 0; v < 8; ++v) {
      const int m = v + 8 * g;
      out[((size_t)b * TT + t0 + m) * HH + hh * 16 + n] = O[hh][v] / rowsum[v];
    }
}

// ---------------------------------------------------------------------------
extern "C" void kernel_launch(void* const* d_in, const int* in_sizes, int n_in,
                              void* d_out, int out_size, void* d_ws, size_t ws_size,
                              hipStream_t stream) {
  (void)in_sizes; (void)n_in; (void)out_size; (void)ws_size;
  const float* x  = (const float*)d_in[0];
  const float* Wq = (const float*)d_in[1];
  const float* Wk = (const float*)d_in[2];
  const float* Wv = (const float*)d_in[3];
  const float* Er = (const float*)d_in[4];
  float* out = (float*)d_out;

  _Float16* qh   = (_Float16*)d_ws;
  _Float16* kh   = qh + (size_t)BB * TT * HH;
  _Float16* vT   = kh + (size_t)BB * TT * HH;
  _Float16* ErH  = vT + (size_t)BB * TT * HH;
  _Float16* Wswz = ErH + (size_t)TT * HH;

  const int prepTot = 3 * CC * HH + TT * HH;
  prep_kernel<<<(prepTot + 255) / 256, 256, 0, stream>>>(Wq, Wk, Wv, Er, Wswz, ErH);
  proj_kernel<<<(BB * TT) / 64, 128, 0, stream>>>(x, Wswz, qh, kh, vT);
  attn_kernel<<<BB * (TT / 64), 128, 0, stream>>>(qh, kh, vT, ErH, out);
}
